// Attention_63556926046483
// MI455X (gfx1250) — matrix operand order repaired and checked by reading, not compiled, not run on hardware
//
#include <hip/hip_runtime.h>
#include <math.h>
#include <stdint.h>

typedef _Float16 h16;
typedef _Float16 v16h __attribute__((ext_vector_type(16)));
typedef _Float16 v8h  __attribute__((ext_vector_type(8)));
typedef float    v8f  __attribute__((ext_vector_type(8)));
typedef unsigned int u32x4 __attribute__((ext_vector_type(4)));
typedef int          i32x8 __attribute__((ext_vector_type(8)));
typedef int          i32x4 __attribute__((ext_vector_type(4)));

#define B_   2
#define S_   2048
#define H_   16
#define KV_  4
#define D_   128
#define GQ_  (H_ / KV_)              // 4
#define QSCALE 0.08838834764831845f // 1/sqrt(128)
#define PSTRIDE 40                   // P staging LDS row stride (halves)
#define KSTR 136                     // K tile LDS row stride (halves): 64dw + 4dw pad
#define VSTR 40                      // V tile LDS row stride (halves): 16dw + 4dw pad

// ---------------------------------------------------------------------------
// DPP16 butterfly reduction over each 16-lane half (VALU only, no LDS)
// ---------------------------------------------------------------------------
template <int CTRL>
__device__ __forceinline__ float dpp_mov(float v) {
    int r = __builtin_amdgcn_update_dpp(0, __builtin_bit_cast(int, v),
                                        CTRL, 0xF, 0xF, true);
    return __builtin_bit_cast(float, r);
}
__device__ __forceinline__ float red16_max(float a) {
    a = fmaxf(a, dpp_mov<0x0B1>(a));  // quad_perm [1,0,3,2]  (xor 1)
    a = fmaxf(a, dpp_mov<0x04E>(a));  // quad_perm [2,3,0,1]  (xor 2)
    a = fmaxf(a, dpp_mov<0x141>(a));  // row_half_mirror      (combine quads)
    a = fmaxf(a, dpp_mov<0x140>(a));  // row_mirror           (combine octets)
    return a;
}
__device__ __forceinline__ float red16_sum(float a) {
    a += dpp_mov<0x0B1>(a);
    a += dpp_mov<0x04E>(a);
    a += dpp_mov<0x141>(a);
    a += dpp_mov<0x140>(a);
    return a;
}

// ---------------------------------------------------------------------------
// TDM: 2D tile load Global -> LDS (data_size = 2 bytes), ISA D# layout.
// g1dw0 carries data_size + pad controls. 6-arg builtin form.
// ---------------------------------------------------------------------------
__device__ __forceinline__ void tdm_load_2d(const void* gptr, unsigned lds_off,
                                            unsigned tensor_d0, unsigned tensor_d1,
                                            unsigned tile_d0, unsigned tile_d1,
                                            unsigned stride0, unsigned g1dw0)
{
    unsigned long long ga = (unsigned long long)(size_t)gptr;
    u32x4 g0;
    g0[0] = 1u;                                    // count=1, user descriptor
    g0[1] = lds_off;                               // lds_addr (bytes)
    g0[2] = (unsigned)ga;                          // global_addr[31:0]
    g0[3] = (unsigned)((ga >> 32) & 0x01FFFFFFull) // global_addr[56:32]
          | (2u << 30);                            // type = 2 ("image")
    i32x8 g1;
    g1[0] = (int)g1dw0;                                          // ds/pad ctl
    g1[1] = (int)((tensor_d0 & 0xFFFFu) << 16);                  // dim0[15:0]
    g1[2] = (int)((tensor_d0 >> 16) | ((tensor_d1 & 0xFFFFu) << 16));
    g1[3] = (int)((tensor_d1 >> 16) | (tile_d0 << 16));
    g1[4] = (int)tile_d1;                                        // tile_dim1 (tile_dim2=0)
    g1[5] = (int)stride0;                                        // dim0_stride[31:0]
    g1[6] = 0;
    g1[7] = 0;
    i32x4 z4 = {0, 0, 0, 0};
    i32x8 z8 = {0, 0, 0, 0, 0, 0, 0, 0};
    __builtin_amdgcn_tensor_load_to_lds(g0, g1, z4, z4, z8, 0);
}

// dw0 of group1: data_size=2B (1<<16), pad_enable, pad_interval, pad_amount(4 dw)
#define G1DW0_K ((1u << 16) | (1u << 20) | (5u << 22) | (3u << 25))  // 64dw rows
#define G1DW0_V ((1u << 16) | (1u << 20) | (3u << 22) | (3u << 25))  // 16dw rows

// ---------------------------------------------------------------------------
// Prep: L2-normalize + RoPE, fp32 -> f16. One 128-thread block per row.
// ---------------------------------------------------------------------------
__global__ __launch_bounds__(128)
void prep_qk_kernel(const float* __restrict__ xq, const float* __restrict__ xk,
                    h16* __restrict__ qh, h16* __restrict__ kh)
{
    __shared__ float red[128];
    __shared__ float xb[128];
    const int t   = threadIdx.x;
    const int row = blockIdx.x;
    const int QROWS = B_ * H_ * S_;

    const float* src;
    h16* dst;
    int s;
    float oscale;
    if (row < QROWS) {
        int b  = row / (H_ * S_);
        int hh = (row / S_) % H_;
        s      = row % S_;
        src = xq + ((size_t)(b * S_ + s)) * (H_ * D_) + (size_t)hh * D_;
        dst = qh + (size_t)row * D_;
        oscale = QSCALE;
    } else {
        int rk = row - QROWS;
        int b  = rk / (KV_ * S_);
        int hh = (rk / S_) % KV_;
        s      = rk % S_;
        src = xk + ((size_t)(b * S_ + s)) * (KV_ * D_) + (size_t)hh * D_;
        dst = kh + (size_t)rk * D_;
        oscale = 1.0f;
    }

    float x = src[t];
    red[t] = x * x;
    __syncthreads();
    for (int off = 64; off >= 1; off >>= 1) {
        if (t < off) red[t] += red[t + off];
        __syncthreads();
    }
    const float nrm = sqrtf(red[0]);
    const float scl = 1.0f / fmaxf(nrm, 1e-6f);
    const float xn  = x * scl;
    xb[t] = xn;
    __syncthreads();

    float y;
    if (t < 64) {
        float inv = powf(1.0f / 10000.0f, (float)t / 64.0f);
        float ang = (float)s * inv;
        y = xn * cosf(ang) + xb[t + 64] * sinf(ang);
    } else {
        int i = t - 64;
        float inv = powf(1.0f / 10000.0f, (float)i / 64.0f);
        float ang = (float)s * inv;
        y = -xb[i] * sinf(ang) + xn * cosf(ang);
    }
    dst[t] = (h16)(y * oscale);
}

// ---------------------------------------------------------------------------
// Prep: V fp32 -> f16 transposed to [b, kv, d, s]
// ---------------------------------------------------------------------------
__global__ __launch_bounds__(256)
void prep_v_kernel(const float* __restrict__ xv, h16* __restrict__ vt)
{
    int idx = blockIdx.x * 256 + threadIdx.x;
    if (idx >= B_ * KV_ * S_ * D_) return;
    int d  = idx % D_;
    int kv = (idx / D_) % KV_;
    int s  = (idx / (D_ * KV_)) % S_;
    int b  = idx / (D_ * KV_ * S_);
    vt[(((size_t)(b * KV_ + kv)) * D_ + d) * S_ + s] = (h16)xv[idx];
}

// ---------------------------------------------------------------------------
// Flash attention. Block = 4 waves = the 4 Q heads of one KV group at one
// 16-row q-tile; K/V tiles TDM-loaded into LDS once and shared by all waves.
// ---------------------------------------------------------------------------
__global__ __launch_bounds__(128)
void fa_kernel(const h16* __restrict__ qh, const h16* __restrict__ kh,
               const h16* __restrict__ vt, float* __restrict__ out)
{
    __shared__ h16 ktile[2][32 * KSTR];    // 32 keys x 128 dims (padded rows)
    __shared__ h16 vtile[2][128 * VSTR];   // 128 dims x 32 keys (padded rows)
    __shared__ h16 pbuf[4][16 * PSTRIDE];  // per-wave P staging

    const int wave = threadIdx.x >> 5;
    const int lane = threadIdx.x & 31;

    const int qt = blockIdx.x % (S_ / 16);
    const int kv = (blockIdx.x / (S_ / 16)) % KV_;
    const int b  = blockIdx.x / ((S_ / 16) * KV_);
    const int h  = kv * GQ_ + wave;
    const int qbase = qt * 16;

    const int lm = lane & 15;
    const int lh = lane >> 4;
    const int kb = lh ? 8 : 0;

    // ---- load Q tile as 4 A-matrices (16x32 each) --------------------------
    const h16* qrow = qh + (((size_t)(b * H_ + h)) * S_ + (qbase + lm)) * D_;
    v16h qa[4];
    for (int c = 0; c < 4; ++c) {
        v8h lo = *(const v8h*)(qrow + c * 32 + kb);
        v8h hi = *(const v8h*)(qrow + c * 32 + 16 + kb);
        for (int i = 0; i < 8; ++i) { qa[c][i] = lo[i]; qa[c][i + 8] = hi[i]; }
    }

    const h16* kbp = kh + ((size_t)(b * KV_ + kv)) * S_ * D_;   // [s][d]
    const h16* vbp = vt + ((size_t)(b * KV_ + kv)) * D_ * S_;   // [d][s]

    v8f oacc[8] = {};
    float mrow[8], lrow[8];
    for (int r = 0; r < 8; ++r) { mrow[r] = -__builtin_inff(); lrow[r] = 0.0f; }

    h16* pl = pbuf[wave];
    const int nblk = (qbase + 16 + 31) >> 5;

    // ---- prologue: TDM-load block 0 into buffer 0 --------------------------
    if (wave == 0) {
        tdm_load_2d(kbp, (unsigned)(size_t)&ktile[0][0],
                    D_, S_, D_, 32, D_, G1DW0_K);
        tdm_load_2d(vbp, (unsigned)(size_t)&vtile[0][0],
                    S_, D_, 32, D_, S_, G1DW0_V);
    }

    for (int blk = 0; blk < nblk; ++blk) {
        const int k0  = blk * 32;
        const int buf = blk & 1;

        if (wave == 0) __builtin_amdgcn_s_wait_tensorcnt(0);
        __syncthreads();   // current buf ready for all waves; other buf idle

        if (wave == 0 && blk + 1 < nblk) {
            const int kn = k0 + 32;
            tdm_load_2d(kbp + (size_t)kn * D_, (unsigned)(size_t)&ktile[buf ^ 1][0],
                        D_, S_ - kn, D_, 32, D_, G1DW0_K);
            tdm_load_2d(vbp + kn, (unsigned)(size_t)&vtile[buf ^ 1][0],
                        S_ - kn, D_, 32, D_, S_, G1DW0_V);
        }

        const h16* kl = ktile[buf];
        const h16* vl = vtile[buf];

        // ---- scores: two 16x16 tiles, K-dim 128 = 4 chunks -----------------
        v8f sc0 = {}, sc1 = {};
        for (int c = 0; c < 4; ++c) {
            v16h b0, b1;
            {
                v8h l0 = *(const v8h*)(kl + (size_t)lm * KSTR        + c * 32 + kb);
                v8h h0 = *(const v8h*)(kl + (size_t)lm * KSTR        + c * 32 + 16 + kb);
                v8h l1 = *(const v8h*)(kl + (size_t)(16 + lm) * KSTR + c * 32 + kb);
                v8h h1 = *(const v8h*)(kl + (size_t)(16 + lm) * KSTR + c * 32 + 16 + kb);
                for (int i = 0; i < 8; ++i) { b0[i] = l0[i]; b0[i + 8] = h0[i]; b1[i] = l1[i]; b1[i + 8] = h1[i]; }
            }
            sc0 = __builtin_amdgcn_wmma_f32_16x16x32_f16(false, qa[c], false, b0,
                                                         (short)0, sc0, false, false);
            sc1 = __builtin_amdgcn_wmma_f32_16x16x32_f16(false, qa[c], false, b1,
                                                         (short)0, sc1, false, false);
        }

        // ---- causal mask (only blocks touching the diagonal) ---------------
        if (k0 + 31 > qbase) {
            for (int r = 0; r < 8; ++r) {
                int q = qbase + r + 8 * lh;
                sc0[r] = (k0 + lm      > q) ? -__builtin_inff() : sc0[r];
                sc1[r] = (k0 + 16 + lm > q) ? -__builtin_inff() : sc1[r];
            }
        }

        // ---- online softmax (DPP reductions, no LDS) -----------------------
        v8f p0, p1;
        float alpha[8];
        for (int r = 0; r < 8; ++r) {
            float a  = red16_max(fmaxf(sc0[r], sc1[r]));
            float mn = fmaxf(mrow[r], a);
            float al = __expf(mrow[r] - mn);
            float e0 = __expf(sc0[r] - mn);
            float e1 = __expf(sc1[r] - mn);
            float sm = red16_sum(e0 + e1);
            lrow[r] = lrow[r] * al + sm;
            mrow[r] = mn;
            alpha[r] = al;
            p0[r] = e0; p1[r] = e1;
        }
        for (int n = 0; n < 8; ++n)
            for (int r = 0; r < 8; ++r) oacc[n][r] *= alpha[r];

        // ---- C-layout -> A-layout via per-wave LDS -------------------------
        for (int r = 0; r < 8; ++r) {
            int rowm = r + 8 * lh;
            pl[rowm * PSTRIDE + lm]      = (h16)p0[r];
            pl[rowm * PSTRIDE + 16 + lm] = (h16)p1[r];
        }
        v16h pa;
        {
            v8h lo = *(const v8h*)(pl + lm * PSTRIDE + kb);
            v8h hi = *(const v8h*)(pl + lm * PSTRIDE + 16 + kb);
            for (int i = 0; i < 8; ++i) { pa[i] = lo[i]; pa[i + 8] = hi[i]; }
        }

        // ---- O += P * V : 8 N-chunks of 16 dims ----------------------------
        for (int n = 0; n < 8; ++n) {
            v16h vb;
            {
                v8h lo = *(const v8h*)(vl + (size_t)(n * 16 + lm) * VSTR + kb);
                v8h hi = *(const v8h*)(vl + (size_t)(n * 16 + lm) * VSTR + 16 + kb);
                for (int i = 0; i < 8; ++i) { vb[i] = lo[i]; vb[i + 8] = hi[i]; }
            }
            oacc[n] = __builtin_amdgcn_wmma_f32_16x16x32_f16(false, pa, false, vb,
                                                             (short)0, oacc[n], false, false);
        }

        __syncthreads();   // all waves done reading buf before it is reloaded
    }

    // ---- epilogue: O /= l, write [b, s, h*D + d] fp32 ----------------------
    float* obase = out + (size_t)b * S_ * (H_ * D_) + (size_t)h * D_;
    for (int r = 0; r < 8; ++r) {
        float inv = 1.0f / lrow[r];
        int s = qbase + r + 8 * lh;
        float* dst = obase + (size_t)s * (H_ * D_);
        for (int n = 0; n < 8; ++n)
            dst[n * 16 + lm] = oacc[n][r] * inv;
    }
}

// ---------------------------------------------------------------------------
extern "C" void kernel_launch(void* const* d_in, const int* in_sizes, int n_in,
                              void* d_out, int out_size, void* d_ws, size_t ws_size,
                              hipStream_t stream)
{
    const float* xq = (const float*)d_in[0];
    const float* xk = (const float*)d_in[1];
    const float* xv = (const float*)d_in[2];
    float* out = (float*)d_out;

    const size_t nq = (size_t)B_ * H_ * S_ * D_;
    const size_t nk = (size_t)B_ * KV_ * S_ * D_;
    h16* qh = (h16*)d_ws;
    h16* kh = qh + nq;
    h16* vt = kh + nk;

    prep_qk_kernel<<<B_ * (H_ + KV_) * S_, 128, 0, stream>>>(xq, xk, qh, kh);
    const int nv = B_ * KV_ * S_ * D_;
    prep_v_kernel<<<(nv + 255) / 256, 256, 0, stream>>>(xv, vt);
    fa_kernel<<<B_ * KV_ * (S_ / 16), 128, 0, stream>>>(qh, kh, vt, out);
}
